// xLSTM_63393717289391
// MI455X (gfx1250) — matrix operand order repaired and checked by reading, not compiled, not run on hardware
//
#include <hip/hip_runtime.h>
#include <stdint.h>

// Problem sizes (fixed by the reference).
#define B_SZ 8
#define T_SZ 2048
#define I_SZ 512
#define H_SZ 8
#define D_SZ 32
#define M_SZ (B_SZ * T_SZ)   // 16384 rows of the big GEMM
#define KC_N (I_SZ / 32)     // 16 K-chunks of 32
#define NPAD 1152            // 1040 real cols padded to 72 n-tiles of 16
#define NT_N (NPAD / 16)     // 72 n-tiles
#define NCOL_O 0
#define NCOL_V 256
#define NCOL_K 512
#define NCOL_Q 768
#define NCOL_F 1024
#define NCOL_I 1032
#define CH 32                // scan: timesteps per async-staged chunk
#define CHUNKS (T_SZ / CH)   // 64

typedef __attribute__((ext_vector_type(16))) __bf16 v16bf;
typedef __attribute__((ext_vector_type(8)))  float v8f;
typedef __attribute__((ext_vector_type(8)))  unsigned int v8u;

static __device__ __forceinline__ unsigned short f32_to_bf16(float f) {
    unsigned int u = __float_as_uint(f);
    unsigned int r = 0x7FFFu + ((u >> 16) & 1u);   // round-to-nearest-even
    return (unsigned short)((u + r) >> 16);
}
static __device__ __forceinline__ unsigned int pack_bf16x2(float lo, float hi) {
    return (unsigned int)f32_to_bf16(lo) | ((unsigned int)f32_to_bf16(hi) << 16);
}

// Raw 32-bit LDS byte offset of a __shared__ object (for async-to-LDS ops).
static __device__ __forceinline__ unsigned lds_off(const void* p) {
    return (unsigned)(unsigned long long)
        (__attribute__((address_space(3))) const char*)p;
}

// Logical fused weight matrix W[k][n], k<512, n<NPAD.
static __device__ __forceinline__ float fetch_w(
    int k, int n,
    const float* __restrict__ Wo, const float* __restrict__ Wf,
    const float* __restrict__ Wi, const float* __restrict__ Wv,
    const float* __restrict__ Wk, const float* __restrict__ Wq)
{
    if (n < 1024) {
        const int grp = n >> 8;          // 0:o 1:v 2:k 3:q
        const int h = (n >> 5) & 7;
        const int j = n & 31;
        const float* W = (grp == 0) ? Wo : (grp == 1) ? Wv : (grp == 2) ? Wk : Wq;
        return W[((size_t)h * I_SZ + k) * D_SZ + j];          // [H,I,D]
    } else if (n < NCOL_F + H_SZ) {
        return Wf[(size_t)(n - NCOL_F) * I_SZ + k];           // [H,I]
    } else if (n < NCOL_I + H_SZ) {
        return Wi[(size_t)(n - NCOL_I) * I_SZ + k];
    }
    return 0.0f;
}

// ---------------------------------------------------------------------------
// A operand: fp32 x -> bf16, pre-swizzled into WMMA A-fragment order.
// Block (mt, kc) = 1KB: lane l holds 16 bf16 =
//   row m = mt*16 + (l&15), K = kc*32 + { 8*hi+[0..8), 16+8*hi+[0..8) }.
// ---------------------------------------------------------------------------
__global__ __launch_bounds__(256) void convert_x_swizzle(
    const float* __restrict__ x, unsigned short* __restrict__ Asw)
{
    const int tid = blockIdx.x * 256 + threadIdx.x;   // exact: M*I/8 threads
    const int d   = tid * 8;                          // dest element index
    const int blk = d >> 9;                           // 1KB block = 512 elems
    const int w   = d & 511;
    const int lane = w >> 4;
    const int e0   = w & 15;                          // 0 or 8
    const int mt = blk >> 4;                          // M/16 tiles
    const int kc = blk & 15;
    const int row = mt * 16 + (lane & 15);
    const int k0  = kc * 32 + (e0 ? 16 : 0) + ((lane >> 4) << 3);
    const float4 f0 = *reinterpret_cast<const float4*>(x + (size_t)row * I_SZ + k0);
    const float4 f1 = *reinterpret_cast<const float4*>(x + (size_t)row * I_SZ + k0 + 4);
    uint4 o;
    o.x = pack_bf16x2(f0.x, f0.y);
    o.y = pack_bf16x2(f0.z, f0.w);
    o.z = pack_bf16x2(f1.x, f1.y);
    o.w = pack_bf16x2(f1.z, f1.w);
    *reinterpret_cast<uint4*>(Asw + d) = o;
}

// ---------------------------------------------------------------------------
// B operand: weights -> bf16, pre-swizzled into WMMA B-fragment order.
// Block (kc, nt) = 1KB: lane l holds 16 bf16 =
//   col n = nt*16 + (l&15), K = kc*32 + { 8*hi+[0..8), 16+8*hi+[0..8) }.
// ---------------------------------------------------------------------------
__global__ __launch_bounds__(256) void pack_weights_swizzle(
    const float* __restrict__ Wo, const float* __restrict__ Wf,
    const float* __restrict__ Wi, const float* __restrict__ Wv,
    const float* __restrict__ Wk, const float* __restrict__ Wq,
    unsigned short* __restrict__ Bsw)
{
    const int tid = blockIdx.x * 256 + threadIdx.x;   // exact: 512*NPAD/2
    const int d   = tid * 2;
    const int blk = d >> 9;
    const int w   = d & 511;
    const int lane = w >> 4;
    const int e    = w & 15;                          // even
    const int kc = blk / NT_N;
    const int nt = blk % NT_N;
    const int k  = kc * 32 + ((e >> 3) << 4) + ((lane >> 4) << 3) + (e & 7);
    const int n  = nt * 16 + (lane & 15);
    const float w0 = fetch_w(k,     n, Wo, Wf, Wi, Wv, Wk, Wq);
    const float w1 = fetch_w(k + 1, n, Wo, Wf, Wi, Wv, Wk, Wq);
    *reinterpret_cast<unsigned int*>(Bsw + d) = pack_bf16x2(w0, w1);
}

__global__ __launch_bounds__(256) void pack_bias(
    const float* __restrict__ bo, const float* __restrict__ bf,
    const float* __restrict__ bi, const float* __restrict__ bv,
    const float* __restrict__ bk, const float* __restrict__ bq,
    float* __restrict__ bc)
{
    const int n = blockIdx.x * 256 + threadIdx.x;
    if (n >= NPAD) return;
    float b = 0.0f;
    if (n < 1024) {
        const int grp = n >> 8;
        const float* bb = (grp == 0) ? bo : (grp == 1) ? bv : (grp == 2) ? bk : bq;
        b = bb[n & 255];                 // h*32 + j
    } else if (n < NCOL_F + H_SZ) {
        b = bf[n - NCOL_F];
    } else if (n < NCOL_I + H_SZ) {
        b = bi[n - NCOL_I];
    }
    bc[n] = b;
}

// ---------------------------------------------------------------------------
// WMMA GEMM, LDS-free. All loads of an iteration (4 B frags + next A frag)
// are issued before the WMMA group so the wait pass can emit partial
// s_wait_loadcnt and overlap loads with matrix math.
// ---------------------------------------------------------------------------
__global__ __launch_bounds__(256) void gemm_wmma_bf16(
    const unsigned short* __restrict__ Asw,
    const unsigned short* __restrict__ Bsw,
    const float* __restrict__ bias,
    float* __restrict__ Y)                  // [M, NPAD] fp32
{
    const int wave = threadIdx.x >> 5;
    const int lane = threadIdx.x & 31;
    const int mt  = blockIdx.x * 8 + wave;  // 16-row tile index
    const int nt0 = blockIdx.y * 4;         // first of 4 n-tiles
    const int hi  = lane >> 4;
    const int l15 = lane & 15;

    const unsigned short* aBase = Asw + ((size_t)mt * KC_N) * 512 + lane * 16;
    const unsigned short* bBase = Bsw + (size_t)nt0 * 512 + lane * 16;

    v8f acc[4] = {};
    v8u aCur = *reinterpret_cast<const v8u*>(aBase);
#pragma unroll 4
    for (int kc = 0; kc < KC_N; ++kc) {
        const v8u b0 = *reinterpret_cast<const v8u*>(bBase + ((size_t)kc * NT_N + 0) * 512);
        const v8u b1 = *reinterpret_cast<const v8u*>(bBase + ((size_t)kc * NT_N + 1) * 512);
        const v8u b2 = *reinterpret_cast<const v8u*>(bBase + ((size_t)kc * NT_N + 2) * 512);
        const v8u b3 = *reinterpret_cast<const v8u*>(bBase + ((size_t)kc * NT_N + 3) * 512);
        v8u aNext = aCur;
        if (kc + 1 < KC_N)
            aNext = *reinterpret_cast<const v8u*>(aBase + (size_t)(kc + 1) * 512);
        const v16bf aF = __builtin_bit_cast(v16bf, aCur);
        acc[0] = __builtin_amdgcn_wmma_f32_16x16x32_bf16(
            false, aF, false, __builtin_bit_cast(v16bf, b0), (short)0, acc[0], false, false);
        acc[1] = __builtin_amdgcn_wmma_f32_16x16x32_bf16(
            false, aF, false, __builtin_bit_cast(v16bf, b1), (short)0, acc[1], false, false);
        acc[2] = __builtin_amdgcn_wmma_f32_16x16x32_bf16(
            false, aF, false, __builtin_bit_cast(v16bf, b2), (short)0, acc[2], false, false);
        acc[3] = __builtin_amdgcn_wmma_f32_16x16x32_bf16(
            false, aF, false, __builtin_bit_cast(v16bf, b3), (short)0, acc[3], false, false);
        aCur = aNext;
    }

    // C/D layout: lane l, element r -> (M = r + 8*hi, N = l&15)
    const int m0 = mt * 16;
#pragma unroll
    for (int nf = 0; nf < 4; ++nf) {
        const int N = (nt0 + nf) * 16 + l15;
        const float bn = bias[N];
#pragma unroll
        for (int r = 0; r < 8; ++r) {
            Y[(size_t)(m0 + r + hi * 8) * NPAD + N] = acc[nf][r] + bn;
        }
    }
}

// ---------------------------------------------------------------------------
// Sequential mLSTM scan, one wave32 per (b,h). Inputs for CH timesteps are
// staged into LDS with async-to-LDS copies, double-buffered: issue chunk c+1,
// then s_wait_asynccnt 18 (= ops per chunk; async loads complete in order, so
// this guarantees chunk c has landed while c+1 streams). Per step everything
// comes from LDS: k row broadcast via 8x ds_load_b128 (no shuffles).
// ---------------------------------------------------------------------------
__global__ __launch_bounds__(32) void mlstm_scan(
    const float* __restrict__ Y, float* __restrict__ out)
{
    __shared__ __align__(16) float kbuf[2][CH][32];
    __shared__ __align__(16) float vbuf[2][CH][32];
    __shared__ float fbuf[2][CH];
    __shared__ float ibuf[2][CH];

    const int bh = blockIdx.x;            // 0..63
    const int b = bh >> 3, h = bh & 7;
    const int lane = threadIdx.x;
    const float kscale = 0.35355339059327373f;   // 1/sqrt(H=8)

    const float* base = Y + (size_t)b * T_SZ * NPAD;
    const int co = h * 32 + lane;

    const int subStep = lane >> 3;        // 4 steps per b128 instruction
    const int seg     = lane & 7;         // 16B segment of a 128B row

    // Issue the 18 async copies (8x k, 8x v as b128; f, i as b32) for the
    // chunk of CH timesteps starting at t0, into buffer `buf`.
    auto issue = [&](int t0, int buf) {
#pragma unroll
        for (int j = 0; j < CH / 4; ++j) {
            const int s = 4 * j + subStep;
            const size_t rowOff = (size_t)(t0 + s) * NPAD;
            const unsigned long long gk =
                (unsigned long long)(base + rowOff + NCOL_K + h * 32) + seg * 16;
            const unsigned long long gv =
                (unsigned long long)(base + rowOff + NCOL_V + h * 32) + seg * 16;
            const unsigned lk = lds_off(&kbuf[buf][s][0]) + seg * 16;
            const unsigned lv = lds_off(&vbuf[buf][s][0]) + seg * 16;
            asm volatile("global_load_async_to_lds_b128 %0, %1, off"
                         :: "v"(lk), "v"(gk) : "memory");
            asm volatile("global_load_async_to_lds_b128 %0, %1, off"
                         :: "v"(lv), "v"(gv) : "memory");
        }
        const size_t rowF = (size_t)(t0 + lane) * NPAD;
        const unsigned long long gf = (unsigned long long)(base + rowF + NCOL_F + h);
        const unsigned long long gi = (unsigned long long)(base + rowF + NCOL_I + h);
        const unsigned lf = lds_off(&fbuf[buf][lane]);
        const unsigned li = lds_off(&ibuf[buf][lane]);
        asm volatile("global_load_async_to_lds_b32 %0, %1, off"
                     :: "v"(lf), "v"(gf) : "memory");
        asm volatile("global_load_async_to_lds_b32 %0, %1, off"
                     :: "v"(li), "v"(gi) : "memory");
    };

    float Crow[32];
#pragma unroll
    for (int c = 0; c < 32; ++c) Crow[c] = 0.0f;
    float n_own = 0.0f, m = 0.0f;

    issue(0, 0);
    for (int c = 0; c < CHUNKS; ++c) {
        const int buf = c & 1;
        if (c + 1 < CHUNKS) {
            issue((c + 1) * CH, buf ^ 1);
            asm volatile("s_wait_asynccnt 0x12" ::: "memory");  // <=18 newest pending
        } else {
            asm volatile("s_wait_asynccnt 0x0" ::: "memory");
        }
#pragma unroll 1
        for (int s = 0; s < CH; ++s) {
            const float fg = fbuf[buf][s];
            const float ig = ibuf[buf][s];
            const float v_own = vbuf[buf][s][lane];
            const float m_new = fmaxf(fg + m, ig);
            const float fe  = __expf(fg + m - m_new);
            const float iek = __expf(ig - m_new) * kscale;  // k-scale folded in
            m = m_new;
            const float iev = iek * v_own;
            const float4* krow = reinterpret_cast<const float4*>(&kbuf[buf][s][0]);
#pragma unroll
            for (int c4 = 0; c4 < 8; ++c4) {
                const float4 kk = krow[c4];
                Crow[4 * c4 + 0] = fmaf(fe, Crow[4 * c4 + 0], iev * kk.x);
                Crow[4 * c4 + 1] = fmaf(fe, Crow[4 * c4 + 1], iev * kk.y);
                Crow[4 * c4 + 2] = fmaf(fe, Crow[4 * c4 + 2], iev * kk.z);
                Crow[4 * c4 + 3] = fmaf(fe, Crow[4 * c4 + 3], iev * kk.w);
            }
            n_own = fmaf(fe, n_own, iek * kbuf[buf][s][lane]);
        }
    }

    // Finalize with the t = T-1 projections.
    const float* last = base + (size_t)(T_SZ - 1) * NPAD;
    const float q_own = last[NCOL_Q + co];
    const float o_own = 1.0f / (1.0f + __expf(-last[NCOL_O + co]));

    float nq = n_own * q_own;
#pragma unroll
    for (int off = 16; off > 0; off >>= 1) nq += __shfl_xor(nq, off, 32);
    const float denom = fmaxf(nq, 1.0f);

    // Broadcast q through LDS for the C*q row dot products.
    kbuf[0][0][lane] = q_own;
    __syncthreads();
    const float4* qrow = reinterpret_cast<const float4*>(&kbuf[0][0][0]);
    float dot = 0.0f;
#pragma unroll
    for (int c4 = 0; c4 < 8; ++c4) {
        const float4 qq = qrow[c4];
        dot = fmaf(Crow[4 * c4 + 0], qq.x, dot);
        dot = fmaf(Crow[4 * c4 + 1], qq.y, dot);
        dot = fmaf(Crow[4 * c4 + 2], qq.z, dot);
        dot = fmaf(Crow[4 * c4 + 3], qq.w, dot);
    }

    // out = [h: B*H*D floats][C: B*H*D*D floats]
    out[(size_t)bh * D_SZ + lane] = o_own * dot / denom;

    float* Cout = out + (size_t)B_SZ * H_SZ * D_SZ
                      + ((size_t)bh * D_SZ + lane) * D_SZ;
#pragma unroll
    for (int c = 0; c < 32; c += 4) {
        *reinterpret_cast<float4*>(Cout + c) =
            make_float4(Crow[c], Crow[c + 1], Crow[c + 2], Crow[c + 3]);
    }
}

// ---------------------------------------------------------------------------
extern "C" void kernel_launch(void* const* d_in, const int* in_sizes, int n_in,
                              void* d_out, int out_size, void* d_ws, size_t ws_size,
                              hipStream_t stream) {
    const float* x  = (const float*)d_in[0];
    const float* Wo = (const float*)d_in[1];
    const float* bo = (const float*)d_in[2];
    const float* Wf = (const float*)d_in[3];
    const float* bf = (const float*)d_in[4];
    const float* Wi = (const float*)d_in[5];
    const float* bi = (const float*)d_in[6];
    const float* Wv = (const float*)d_in[7];
    const float* bv = (const float*)d_in[8];
    const float* Wk = (const float*)d_in[9];
    const float* bk = (const float*)d_in[10];
    const float* Wq = (const float*)d_in[11];
    const float* bq = (const float*)d_in[12];

    char* ws = (char*)d_ws;
    size_t off = 0;
    unsigned short* Asw = (unsigned short*)(ws + off); off += (size_t)M_SZ * I_SZ * 2; // 16 MiB
    unsigned short* Bsw = (unsigned short*)(ws + off); off += (size_t)I_SZ * NPAD * 2; // 1.1 MiB
    float* bc           = (float*)(ws + off);          off += (size_t)NPAD * 4;        // 4.5 KiB
    float* Y            = (float*)(ws + off);          off += (size_t)M_SZ * NPAD * 4; // 72 MiB

    convert_x_swizzle<<<(M_SZ * I_SZ / 8) / 256, 256, 0, stream>>>(x, Asw);
    pack_weights_swizzle<<<(I_SZ * NPAD / 2) / 256, 256, 0, stream>>>(
        Wo, Wf, Wi, Wv, Wk, Wq, Bsw);
    pack_bias<<<(NPAD + 255) / 256, 256, 0, stream>>>(bo, bf, bi, bv, bk, bq, bc);

    dim3 gg(M_SZ / (16 * 8), NPAD / 64);   // 128 x 18 blocks, 256 threads
    gemm_wmma_bf16<<<gg, 256, 0, stream>>>(Asw, Bsw, bc, Y);

    mlstm_scan<<<B_SZ * H_SZ, 32, 0, stream>>>(Y, (float*)d_out);
}
